// GraphConv_8847632629923
// MI455X (gfx1250) — matrix operand order repaired and checked by reading, not compiled, not run on hardware
//
#include <hip/hip_runtime.h>
#include <math.h>

// ---------------- problem constants (match reference) ----------------
constexpr int   BGR   = 256;            // graphs
constexpr int   S     = 200;            // NROIS (tokens per graph)
constexpr int   INW   = 200;            // input feature width
constexpr int   HID   = 128;
constexpr int   LNUM  = 4;
constexpr int   NH    = 8;              // heads
constexpr int   KTOP  = 100;            // SAG top-k
constexpr int   NN    = BGR * S;        // 51200 nodes
constexpr int   EE    = BGR * S * 32;   // 1,638,400 edges
constexpr int   EMB   = HID * LNUM;     // 512
constexpr int   DH    = EMB / NH;       // 64

typedef __attribute__((ext_vector_type(16))) _Float16 v16h;
typedef __attribute__((ext_vector_type(8)))  float    v8f;

// ---------------- generic fill ----------------
__global__ void k_fill(float* p, size_t n, float v) {
    size_t i  = (size_t)blockIdx.x * blockDim.x + threadIdx.x;
    size_t st = (size_t)gridDim.x * blockDim.x;
    for (; i < n; i += st) p[i] = v;
}

// ---------------- degree / rsqrt ----------------
__global__ void k_edge_deg(const int* dst, float* deg) {
    int e = blockIdx.x * blockDim.x + threadIdx.x;
    if (e < EE) atomicAdd(&deg[dst[e]], 1.0f);
}
__global__ void k_dinv(const float* deg, float* dinv) {
    int n = blockIdx.x * blockDim.x + threadIdx.x;
    if (n < NN) dinv[n] = rsqrtf(deg[n]);   // deg already includes +1 self loop
}

// ================= WMMA GEMM =================
// C = act(A[M,K] * op(B)[K,N] + bias).  REQUIRES: M % 64 == 0, N % 64 == 0,
// lda/ldb multiples of 4 (float4 alignment). K arbitrary (masked tail).
// Block: 128 threads = 4 waves. Block tile: 64(M) x 64(N).
// Wave w owns rows m0 = blockIdx.y*64 + w*16 and all four 16-wide N subtiles.
// B tile (32K x 64N) staged in LDS in fragment-major f16 layout, shared by waves.

__device__ inline v16h a_frag_full(const float* __restrict__ A, int lda,
                                   int arow, int k0, int half) {
    const float* p = A + (size_t)arow * lda + k0 + half * 8;
    float4 a0 = *(const float4*)(p + 0);
    float4 a1 = *(const float4*)(p + 4);
    float4 a2 = *(const float4*)(p + 16);
    float4 a3 = *(const float4*)(p + 20);
    v16h f;
    f[0]=(_Float16)a0.x;  f[1]=(_Float16)a0.y;  f[2]=(_Float16)a0.z;  f[3]=(_Float16)a0.w;
    f[4]=(_Float16)a1.x;  f[5]=(_Float16)a1.y;  f[6]=(_Float16)a1.z;  f[7]=(_Float16)a1.w;
    f[8]=(_Float16)a2.x;  f[9]=(_Float16)a2.y;  f[10]=(_Float16)a2.z; f[11]=(_Float16)a2.w;
    f[12]=(_Float16)a3.x; f[13]=(_Float16)a3.y; f[14]=(_Float16)a3.z; f[15]=(_Float16)a3.w;
    return f;
}

__device__ inline v16h a_frag_tail(const float* __restrict__ A, int lda,
                                   int arow, int k0, int half, int K) {
    const float* rowp = A + (size_t)arow * lda;
    v16h f;
#pragma unroll
    for (int i = 0; i < 16; ++i) {
        int rk = (i & 7) + ((i >> 3) << 4) + half * 8;
        int kk = k0 + rk;
        int kc = kk < K ? kk : K - 1;        // clamp: unconditional load
        float v = rowp[kc];
        f[i] = (_Float16)(kk < K ? v : 0.0f); // v_cndmask, no EXEC branch
    }
    return f;
}

__global__ void __launch_bounds__(128)
k_wmma_gemm(const float* __restrict__ A, int lda,
            const float* __restrict__ Bm, int ldb, int transB,
            const float* __restrict__ bias,
            float* __restrict__ C, int ldc,
            int M, int Nn, int K, int act) {
    __shared__ _Float16 ldsB[4 * 32 * 16];   // [subtile][lane][elem] = 4KB

    const int tid  = threadIdx.x;
    const int wave = tid >> 5;
    const int lane = tid & 31;
    const int half = lane >> 4;
    const int l16  = lane & 15;

    const int m0 = blockIdx.y * 64 + wave * 16;
    const int n0 = blockIdx.x * 64;
    const int arow = m0 + l16;

    // staging role of this thread: fragment for subtile snt, lane slf
    const int snt  = tid >> 5;
    const int slf  = tid & 31;
    const int skoff = (slf & 16) >> 1;       // K offset 0 or 8
    const int scol  = n0 + (snt << 4) + (slf & 15);

    v8f acc0 = {}, acc1 = {}, acc2 = {}, acc3 = {};

    const int kmain = K & ~31;
    for (int k0 = 0; k0 < kmain; k0 += 32) {
        // ---- stage B tile into LDS (f16, fragment-major) ----
        v16h bf;
        if (transB) {
            const float* bp = Bm + (size_t)scol * ldb + k0 + skoff;
            float4 b0 = *(const float4*)(bp + 0);
            float4 b1 = *(const float4*)(bp + 4);
            float4 b2 = *(const float4*)(bp + 16);
            float4 b3 = *(const float4*)(bp + 20);
            bf[0]=(_Float16)b0.x;  bf[1]=(_Float16)b0.y;  bf[2]=(_Float16)b0.z;  bf[3]=(_Float16)b0.w;
            bf[4]=(_Float16)b1.x;  bf[5]=(_Float16)b1.y;  bf[6]=(_Float16)b1.z;  bf[7]=(_Float16)b1.w;
            bf[8]=(_Float16)b2.x;  bf[9]=(_Float16)b2.y;  bf[10]=(_Float16)b2.z; bf[11]=(_Float16)b2.w;
            bf[12]=(_Float16)b3.x; bf[13]=(_Float16)b3.y; bf[14]=(_Float16)b3.z; bf[15]=(_Float16)b3.w;
        } else {
            const float* bp = Bm + (size_t)(k0 + skoff) * ldb + scol;
#pragma unroll
            for (int i = 0; i < 16; ++i) bf[i] = (_Float16)bp[(size_t)((i & 7) + ((i >> 3) << 4)) * ldb];
        }
        *(v16h*)&ldsB[tid * 16] = bf;

        // ---- A fragment (reused across 4 subtiles) ----
        v16h af = a_frag_full(A, lda, arow, k0, half);
        __builtin_prefetch(A + (size_t)arow * lda + k0 + 32, 0, 3);
        __syncthreads();

        v16h bf0 = *(v16h*)&ldsB[(0 * 32 + lane) * 16];
        v16h bf1 = *(v16h*)&ldsB[(1 * 32 + lane) * 16];
        v16h bf2 = *(v16h*)&ldsB[(2 * 32 + lane) * 16];
        v16h bf3 = *(v16h*)&ldsB[(3 * 32 + lane) * 16];
        acc0 = __builtin_amdgcn_wmma_f32_16x16x32_f16(false, af, false, bf0, (short)0, acc0, false, false);
        acc1 = __builtin_amdgcn_wmma_f32_16x16x32_f16(false, af, false, bf1, (short)0, acc1, false, false);
        acc2 = __builtin_amdgcn_wmma_f32_16x16x32_f16(false, af, false, bf2, (short)0, acc2, false, false);
        acc3 = __builtin_amdgcn_wmma_f32_16x16x32_f16(false, af, false, bf3, (short)0, acc3, false, false);
        __syncthreads();
    }

    if (K & 31) {                            // masked tail (uniform branch)
        const int k0 = kmain;
        v16h bf;
        if (transB) {
            const float* bp = Bm + (size_t)scol * ldb;
#pragma unroll
            for (int i = 0; i < 16; ++i) {
                int kk = k0 + skoff + (i & 7) + ((i >> 3) << 4);
                int kc = kk < K ? kk : K - 1;
                float v = bp[kc];
                bf[i] = (_Float16)(kk < K ? v : 0.0f);
            }
        } else {
#pragma unroll
            for (int i = 0; i < 16; ++i) {
                int kk = k0 + skoff + (i & 7) + ((i >> 3) << 4);
                int kc = kk < K ? kk : K - 1;
                float v = Bm[(size_t)kc * ldb + scol];
                bf[i] = (_Float16)(kk < K ? v : 0.0f);
            }
        }
        *(v16h*)&ldsB[tid * 16] = bf;
        v16h af = a_frag_tail(A, lda, arow, k0, half, K);
        __syncthreads();
        v16h bf0 = *(v16h*)&ldsB[(0 * 32 + lane) * 16];
        v16h bf1 = *(v16h*)&ldsB[(1 * 32 + lane) * 16];
        v16h bf2 = *(v16h*)&ldsB[(2 * 32 + lane) * 16];
        v16h bf3 = *(v16h*)&ldsB[(3 * 32 + lane) * 16];
        acc0 = __builtin_amdgcn_wmma_f32_16x16x32_f16(false, af, false, bf0, (short)0, acc0, false, false);
        acc1 = __builtin_amdgcn_wmma_f32_16x16x32_f16(false, af, false, bf1, (short)0, acc1, false, false);
        acc2 = __builtin_amdgcn_wmma_f32_16x16x32_f16(false, af, false, bf2, (short)0, acc2, false, false);
        acc3 = __builtin_amdgcn_wmma_f32_16x16x32_f16(false, af, false, bf3, (short)0, acc3, false, false);
    }

    // ---- store 4 subtiles ----
    v8f accs[4] = {acc0, acc1, acc2, acc3};
#pragma unroll
    for (int nt = 0; nt < 4; ++nt) {
        int col = n0 + nt * 16 + l16;
        float bv = bias ? bias[col] : 0.0f;
#pragma unroll
        for (int r = 0; r < 8; ++r) {
            int row = m0 + r + half * 8;
            float v = accs[nt][r] + bv;
            if (act) v = tanhf(v);
            C[(size_t)row * ldc + col] = v;
        }
    }
}

// ---------------- edge scatter: out[dst] += feat[src] * norm ----------------
__global__ void k_scatter(const float* __restrict__ feat,
                          const int* __restrict__ src, const int* __restrict__ dst,
                          const float* __restrict__ dinv, int useNorm,
                          float* __restrict__ out) {
    int e = blockIdx.x;
    int f = threadIdx.x;
    int sN = src[e], dN = dst[e];
    float norm = useNorm ? dinv[sN] * dinv[dN] : 1.0f;
    atomicAdd(&out[(size_t)dN * HID + f], feat[(size_t)sN * HID + f] * norm);
}

// ---------------- x1 = tanh(agg + xw*dinv^2 + b) (in-place on agg ok) -------
__global__ void k_selfloop_tanh(const float* __restrict__ agg,
                                const float* __restrict__ xw,
                                const float* __restrict__ dinv,
                                const float* __restrict__ b,
                                float* __restrict__ out) {
    size_t idx = (size_t)blockIdx.x * blockDim.x + threadIdx.x;
    if (idx >= (size_t)NN * HID) return;
    int n = (int)(idx / HID), f = (int)(idx % HID);
    float di = dinv[n];
    out[idx] = tanhf(agg[idx] + xw[idx] * di * di + b[f]);
}

// ---------------- cat = [z | x1] ----------------
__global__ void k_concat(const float* __restrict__ z, int zw,
                         const float* __restrict__ x1,
                         float* __restrict__ cat, int W) {
    size_t idx = (size_t)blockIdx.x * blockDim.x + threadIdx.x;
    if (idx >= (size_t)NN * W) return;
    int n = (int)(idx / W), c = (int)(idx % W);
    cat[idx] = (c < zw) ? z[(size_t)n * zw + c] : x1[(size_t)n * HID + (c - zw)];
}

// ---------------- SAG score: neigh@w1 + b + z@w2 ----------------
__global__ void k_score(const float* __restrict__ neigh, const float* __restrict__ z,
                        const float* __restrict__ w1, const float* __restrict__ w2,
                        const float* __restrict__ sb, float* __restrict__ score) {
    int n = blockIdx.x * blockDim.x + threadIdx.x;
    if (n >= NN) return;
    float s = sb[0];
    const float* np = neigh + (size_t)n * HID;
    const float* zp = z + (size_t)n * HID;
#pragma unroll 4
    for (int f = 0; f < HID; ++f) s += np[f] * w1[f] + zp[f] * w2[f];
    score[n] = s;
}

// ---------------- per-node top-k rank -> scale = keep ? tanh(score) : 0 -----
__global__ void k_sag_scale(const float* __restrict__ score, float* __restrict__ scl) {
    int n = blockIdx.x * blockDim.x + threadIdx.x;
    if (n >= NN) return;
    int b = n / S, base = b * S, li = n - base;
    float my = score[n];
    int rank = 0;
    for (int m = 0; m < S; ++m) {
        float sm = score[base + m];
        rank += (sm > my) || (sm == my && m < li);
    }
    scl[n] = (rank < KTOP) ? tanhf(my) : 0.0f;
}

// ---------------- pads written into xc[:, layer*HID : (layer+1)*HID] --------
__global__ void k_write_xc(const float* __restrict__ z, const float* __restrict__ scl,
                           float* __restrict__ xc, int layer) {
    size_t idx = (size_t)blockIdx.x * blockDim.x + threadIdx.x;
    if (idx >= (size_t)NN * HID) return;
    int n = (int)(idx / HID), f = (int)(idx % HID);
    xc[(size_t)n * EMB + layer * HID + f] = z[idx] * scl[n];
}

// ---------------- attention: one block per (q-row, head, graph) ----------------
__global__ void k_attention(const float* __restrict__ qkv,
                            float* __restrict__ ohd, float* __restrict__ attn) {
    __shared__ float sc[S];
    __shared__ float red;
    int qi = blockIdx.x, h = blockIdx.y, b = blockIdx.z, t = threadIdx.x;
    const size_t rowq = (size_t)(b * S + qi) * (3 * EMB);
    const float* qp = qkv + rowq + h * DH;

    for (int j = t; j < S; j += blockDim.x) {
        const float* kp = qkv + (size_t)(b * S + j) * (3 * EMB) + EMB + h * DH;
        float a = 0.0f;
#pragma unroll 4
        for (int d = 0; d < DH; ++d) a += qp[d] * kp[d];
        sc[j] = a * 0.125f;                       // 1/sqrt(64)
    }
    __syncthreads();
    if (t == 0) {
        float mx = -3.4e38f;
        for (int j = 0; j < S; ++j) mx = fmaxf(mx, sc[j]);
        red = mx;
    }
    __syncthreads();
    float mx = red;
    for (int j = t; j < S; j += blockDim.x) sc[j] = expf(sc[j] - mx);
    __syncthreads();
    if (t == 0) {
        float sm = 0.0f;
        for (int j = 0; j < S; ++j) sm += sc[j];
        red = sm;
    }
    __syncthreads();
    float inv = 1.0f / red;
    for (int j = t; j < S; j += blockDim.x) {
        sc[j] *= inv;
        atomicAdd(&attn[((size_t)b * S + qi) * S + j], sc[j] * (1.0f / NH));
    }
    __syncthreads();
    if (t < DH) {
        float a = 0.0f;
        for (int j = 0; j < S; ++j)
            a += sc[j] * qkv[(size_t)(b * S + j) * (3 * EMB) + 2 * EMB + h * DH + t];
        ohd[(size_t)(b * S + qi) * EMB + h * DH + t] = a;
    }
}

// ---------------- mean over tokens: pooled[b,e] ----------------
__global__ void k_meanpool(const float* __restrict__ o, float* __restrict__ pooled) {
    int idx = blockIdx.x * blockDim.x + threadIdx.x;
    if (idx >= BGR * EMB) return;
    int b = idx / EMB, e = idx % EMB;
    float a = 0.0f;
    for (int s = 0; s < S; ++s) a += o[(size_t)(b * S + s) * EMB + e];
    pooled[idx] = a * (1.0f / S);
}

// ---------------- host-side helpers ----------------
static inline void gemm(hipStream_t st, const float* A, int lda,
                        const float* B, int ldb, int transB, const float* bias,
                        float* C, int ldc, int M, int Nn, int K, int act) {
    // requires M % 64 == 0 and N % 64 == 0 (true at all call sites)
    dim3 g(Nn / 64, M / 64);
    k_wmma_gemm<<<g, 128, 0, st>>>(A, lda, B, ldb, transB, bias, C, ldc, M, Nn, K, act);
}
static inline void fill(hipStream_t st, float* p, size_t n, float v) {
    k_fill<<<4096, 256, 0, st>>>(p, n, v);
}

extern "C" void kernel_launch(void* const* d_in, const int* in_sizes, int n_in,
                              void* d_out, int out_size, void* d_ws, size_t ws_size,
                              hipStream_t stream) {
    const float* x       = (const float*)d_in[0];
    const int*   ei      = (const int*)d_in[1];
    const int*   srcI    = ei;
    const int*   dstI    = ei + EE;
    const float* gcn_w0  = (const float*)d_in[4];
    const float* gcn_w   = (const float*)d_in[5];
    const float* gcn_b   = (const float*)d_in[6];
    const float* lin1_w0 = (const float*)d_in[7];
    const float* lin1_w  = (const float*)d_in[8];
    const float* lin1_b  = (const float*)d_in[9];
    const float* lin2_w  = (const float*)d_in[10];
    const float* lin2_b  = (const float*)d_in[11];
    const float* sag_w1  = (const float*)d_in[12];
    const float* sag_w2  = (const float*)d_in[13];
    const float* sag_b   = (const float*)d_in[14];
    const float* mha_in_w  = (const float*)d_in[15];
    const float* mha_in_b  = (const float*)d_in[16];
    const float* mha_out_w = (const float*)d_in[17];
    const float* mha_out_b = (const float*)d_in[18];
    const float* fin_w   = (const float*)d_in[19];
    const float* fin_b   = (const float*)d_in[20];

    float* outp  = (float*)d_out;               // [256,128]
    float* attnp = outp + (size_t)BGR * HID;    // [256,200,200]

    // ---- workspace carve ----
    float* w = (float*)d_ws;
    size_t off = 0;
    auto carve = [&](size_t n) { float* p = w + off; off += n; return p; };
    float* dinv   = carve(NN);
    float* scl    = carve(NN);
    float* score  = carve(NN);                      // also degree scratch
    float* zbuf   = carve((size_t)NN * HID);
    float* t0     = carve((size_t)NN * HID);        // xw / h
    float* t1     = carve((size_t)NN * HID);        // agg / x1 / neigh
    float* cat    = carve((size_t)NN * (INW + HID));
    float* xc     = carve((size_t)NN * EMB);
    float* qkvb   = carve((size_t)NN * 3 * EMB);    // later reused for out-proj
    float* ohd    = carve((size_t)NN * EMB);
    float* pooled = carve((size_t)BGR * EMB);
    (void)ws_size; (void)n_in; (void)in_sizes; (void)out_size;

    // ---- degree -> dinv (self loop included as initial 1.0) ----
    fill(stream, score, NN, 1.0f);
    k_edge_deg<<<(EE + 255) / 256, 256, 0, stream>>>(dstI, score);
    k_dinv<<<(NN + 255) / 256, 256, 0, stream>>>(score, dinv);

    // ---- 4 GC blocks + SAG pads ----
    const float* z = x;
    int zw = INW;
    const size_t NH_ = (size_t)NN * HID;
    for (int i = 0; i < LNUM; ++i) {
        const float* gw  = (i == 0) ? gcn_w0  : gcn_w  + (size_t)(i - 1) * HID * HID;
        const float* l1w = (i == 0) ? lin1_w0 : lin1_w + (size_t)(i - 1) * 2 * HID * HID;
        const int    l1k = (i == 0) ? (INW + HID) : (2 * HID);
        const float* gb  = gcn_b  + (size_t)i * HID;
        const float* l1b = lin1_b + (size_t)i * HID;
        const float* l2w = lin2_w + (size_t)i * HID * HID;
        const float* l2b = lin2_b + (size_t)i * HID;

        // xw = z @ gw                               [NN,HID]
        gemm(stream, z, zw, gw, HID, 0, nullptr, t0, HID, NN, HID, zw, 0);
        // agg = segsum(xw[src]*dinv[s]*dinv[d] -> dst)
        fill(stream, t1, NH_, 0.0f);
        k_scatter<<<EE, HID, 0, stream>>>(t0, srcI, dstI, dinv, 1, t1);
        // x1 = tanh(agg + xw*dinv^2 + gb)           (in place in t1)
        k_selfloop_tanh<<<(int)((NH_ + 255) / 256), 256, 0, stream>>>(t1, t0, dinv, gb, t1);
        // cat = [z | x1]
        k_concat<<<(int)(((size_t)NN * l1k + 255) / 256), 256, 0, stream>>>(z, zw, t1, cat, l1k);
        // h = tanh(cat @ l1w + l1b)                 [NN,HID] -> t0
        gemm(stream, cat, l1k, l1w, HID, 0, l1b, t0, HID, NN, HID, l1k, 1);
        // z' = tanh(h @ l2w + l2b)                  [NN,HID] -> zbuf
        gemm(stream, t0, HID, l2w, HID, 0, l2b, zbuf, HID, NN, HID, HID, 1);

        // SAG: neigh = segsum(z'[src] -> dst)
        fill(stream, t1, NH_, 0.0f);
        k_scatter<<<EE, HID, 0, stream>>>(zbuf, srcI, dstI, dinv, 0, t1);
        k_score<<<(NN + 127) / 128, 128, 0, stream>>>(t1, zbuf, sag_w1, sag_w2, sag_b, score);
        k_sag_scale<<<(NN + 127) / 128, 128, 0, stream>>>(score, scl);
        k_write_xc<<<(int)((NH_ + 255) / 256), 256, 0, stream>>>(zbuf, scl, xc, i);

        z = zbuf; zw = HID;
    }

    // ---- MHA ----
    // qkv = xc @ mha_in_w^T + mha_in_b              [NN, 3*EMB]
    gemm(stream, xc, EMB, mha_in_w, EMB, 1, mha_in_b, qkvb, 3 * EMB, NN, 3 * EMB, EMB, 0);
    fill(stream, attnp, (size_t)BGR * S * S, 0.0f);
    k_attention<<<dim3(S, NH, BGR), 128, 0, stream>>>(qkvb, ohd, attnp);
    // oproj = ohd @ mha_out_w^T + mha_out_b         [NN, EMB]  (reuse qkv region)
    gemm(stream, ohd, EMB, mha_out_w, EMB, 1, mha_out_b, qkvb, EMB, NN, EMB, EMB, 0);

    // ---- readout ----
    k_meanpool<<<(BGR * EMB + 255) / 256, 256, 0, stream>>>(qkvb, pooled);
    // out = tanh(pooled @ fin_w + fin_b)            [256,128]
    gemm(stream, pooled, EMB, fin_w, HID, 0, fin_b, outp, HID, BGR, HID, EMB, 1);
}
